// Model_59347858096499
// MI455X (gfx1250) — matrix operand order repaired and checked by reading, not compiled, not run on hardware
//
#include <hip/hip_runtime.h>
#include <math.h>

// ---------------------------------------------------------------------------
// Model dims (from reference): B=256, T=600 -> T5=120 steps, K=10, U=128,
// C=128, H=400, OSC=5, out width = 1+6*20 = 121.
// ---------------------------------------------------------------------------
#define BATCH 256
#define TSTEPS 120
#define HDIM  400
#define CDIM  128
#define UDIM  128
#define KATT  10
#define NOUT  121

// Padded K dims (multiples of 32) for WMMA
#define K_R1  544   // [x(3), w(128), h1(400)] = 531 -> 544
#define K_R2  960   // [x(3), h1(400), w(128), h2(400)] = 931 -> 960
#define K_SM  800   // [h2(400), s(400)] = 800 (exact)
#define K_W   416   // [h1(400)] -> 416 ; also z rows padded to 416

typedef _Float16 half_t;
typedef __attribute__((ext_vector_type(16))) _Float16 v16h;
typedef __attribute__((ext_vector_type(8)))  float    v8f;

// ---------------------------------------------------------------------------
// WMMA fragment loaders (wave32, 16x16x32 f16, layouts per CDNA5 ISA 7.12.2)
// A (16x32, MxK): lanes 0-15 row M=lane, K = {k0..k0+7, k0+16..k0+23};
//                 lanes 16-31 same rows, K halves shifted by +8.
// B (32x16, KxN): lane n = lane&15 holds column n; K same as A
//                 ({k0..k0+7, k0+16..k0+23}, +8 for lanes 16-31) -- B sourced from W^T (N x Kpad).
// ---------------------------------------------------------------------------
__device__ __forceinline__ v16h load_fragA(const half_t* __restrict__ base, int ld,
                                           int m0, int k0, int lane) {
    int r  = m0 + (lane & 15);
    int kh = (lane >> 4) * 8;
    const half_t* p = base + (size_t)r * ld + k0 + kh;
    union { v16h v; uint4 q[2]; } u;
    u.q[0] = *(const uint4*)(p);
    u.q[1] = *(const uint4*)(p + 16);
    return u.v;
}

__device__ __forceinline__ v16h load_fragB(const half_t* __restrict__ base, int ld,
                                           int n0, int k0, int lane) {
    int n  = n0 + (lane & 15);
    int kh = (lane >> 4) * 8;
    const half_t* p = base + (size_t)n * ld + k0 + kh;
    union { v16h v; uint4 q[2]; } u;
    u.q[0] = *(const uint4*)(p);
    u.q[1] = *(const uint4*)(p + 16);
    return u.v;
}

__device__ __forceinline__ float sigm(float x) { return 1.0f / (1.0f + expf(-x)); }

// ---------------------------------------------------------------------------
// Workspace zero (deterministic state each call)
// ---------------------------------------------------------------------------
__global__ void zero_ws_kernel(uint4* p, size_t n16) {
    size_t i = (size_t)blockIdx.x * blockDim.x + threadIdx.x;
    if (i < n16) { uint4 z; z.x = z.y = z.z = z.w = 0u; p[i] = z; }
}

// ---------------------------------------------------------------------------
// Weight packing: combined gate weight W^T, shape (800 x Kpad) f16.
// Row n: gate = n/400 (0=j,1=k), col = n%400.
// k-range: [0,dinX) -> input weight rows, [dinX,dinX+400) -> hidden weight.
// ---------------------------------------------------------------------------
__global__ void pack_ff_W_kernel(const float* __restrict__ jxW, const float* __restrict__ jhW,
                                 const float* __restrict__ kxW, const float* __restrict__ khW,
                                 int dinX, half_t* __restrict__ dst, int Kpad) {
    int i = blockIdx.x * blockDim.x + threadIdx.x;
    int total = 2 * HDIM * Kpad;
    if (i >= total) return;
    int n = i / Kpad, k = i % Kpad;
    int gate = n / HDIM;
    int col  = n % HDIM;
    const float* X  = gate ? kxW : jxW;
    const float* Hm = gate ? khW : jhW;
    float v = 0.0f;
    if (k < dinX)             v = X[(size_t)k * HDIM + col];
    else if (k < dinX + HDIM) v = Hm[(size_t)(k - dinX) * HDIM + col];
    dst[i] = (half_t)v;
}

__global__ void pack_ff_bias_kernel(const float* jxb, const float* jhb,
                                    const float* kxb, const float* khb, float* dst) {
    int i = blockIdx.x * blockDim.x + threadIdx.x;
    if (i >= 2 * HDIM) return;
    dst[i] = (i < HDIM) ? (jxb[i] + jhb[i]) : (kxb[i - HDIM] + khb[i - HDIM]);
}

__global__ void pack_win_W_kernel(const float* __restrict__ winW, half_t* __restrict__ dst) {
    int i = blockIdx.x * blockDim.x + threadIdx.x;  // 32*416
    if (i >= 32 * K_W) return;
    int n = i / K_W, k = i % K_W;
    float v = (n < 3 * KATT && k < HDIM) ? winW[(size_t)k * (3 * KATT) + n] : 0.0f;
    dst[i] = (half_t)v;
}

__global__ void pack_lin_W_kernel(const float* __restrict__ linW, half_t* __restrict__ dst) {
    int i = blockIdx.x * blockDim.x + threadIdx.x;  // 128*416
    if (i >= 128 * K_W) return;
    int n = i / K_W, k = i % K_W;
    float v = (n < NOUT && k < HDIM) ? linW[(size_t)k * NOUT + n] : 0.0f;
    dst[i] = (half_t)v;
}

__global__ void pad_vec_kernel(const float* src, int nsrc, float* dst, int ndst) {
    int i = blockIdx.x * blockDim.x + threadIdx.x;
    if (i >= ndst) return;
    dst[i] = (i < nsrc) ? src[i] : 0.0f;
}

// ---------------------------------------------------------------------------
// A-matrix assembly (f32 state -> padded f16 GEMM operand)
// ---------------------------------------------------------------------------
__global__ void assemble_A1_kernel(const float* __restrict__ x, const float* __restrict__ w,
                                   const float* __restrict__ h1, half_t* __restrict__ A, int t) {
    int i = blockIdx.x * blockDim.x + threadIdx.x;  // 256*544
    if (i >= BATCH * K_R1) return;
    int b = i / K_R1, k = i % K_R1;
    float v = 0.0f;
    if (k < 3)        v = x[((size_t)b * 600 + 5 * t) * 3 + k];
    else if (k < 131) v = w[(size_t)b * CDIM + (k - 3)];
    else if (k < 531) v = h1[(size_t)b * HDIM + (k - 131)];
    A[i] = (half_t)v;
}

__global__ void assemble_A2_kernel(const float* __restrict__ x, const float* __restrict__ h1,
                                   const float* __restrict__ w, const float* __restrict__ h2,
                                   half_t* __restrict__ A, int t) {
    int i = blockIdx.x * blockDim.x + threadIdx.x;  // 256*960
    if (i >= BATCH * K_R2) return;
    int b = i / K_R2, k = i % K_R2;
    float v = 0.0f;
    if (k < 3)        v = x[((size_t)b * 600 + 5 * t) * 3 + k];
    else if (k < 403) v = h1[(size_t)b * HDIM + (k - 3)];
    else if (k < 531) v = w[(size_t)b * CDIM + (k - 403)];
    else if (k < 931) v = h2[(size_t)b * HDIM + (k - 531)];
    A[i] = (half_t)v;
}

__global__ void assemble_As_kernel(const float* __restrict__ h2, const float* __restrict__ s,
                                   half_t* __restrict__ A) {
    int i = blockIdx.x * blockDim.x + threadIdx.x;  // 256*800
    if (i >= BATCH * K_SM) return;
    int b = i / K_SM, k = i % K_SM;
    float v = (k < HDIM) ? h2[(size_t)b * HDIM + k] : s[(size_t)b * HDIM + (k - HDIM)];
    A[i] = (half_t)v;
}

__global__ void assemble_Aw_kernel(const float* __restrict__ h1, half_t* __restrict__ A) {
    int i = blockIdx.x * blockDim.x + threadIdx.x;  // 256*416
    if (i >= BATCH * K_W) return;
    int b = i / K_W, k = i % K_W;
    A[i] = (half_t)((k < HDIM) ? h1[(size_t)b * HDIM + k] : 0.0f);
}

// ---------------------------------------------------------------------------
// Fused gated-FF GEMM: pre = A @ Wt (both gates), epilogue:
//   h = sigm(pre_j + bj) * (1-h) + (1 - sigm(pre_k + bk)) * h   (in place)
// 400 tiles = 16(M) x 25(N); launch 100 blocks x 128 threads (4 waves/block).
// Early-exit guard is wave-uniform, so EXEC is all-ones for every WMMA.
// ---------------------------------------------------------------------------
__global__ void ff_gemm_kernel(const half_t* __restrict__ A, int Kpad,
                               const half_t* __restrict__ W, const float* __restrict__ bias,
                               float* __restrict__ h) {
    int lane = threadIdx.x & 31;
    int wave = threadIdx.x >> 5;
    int tile = blockIdx.x * 4 + wave;          // 0..399
    if (tile >= 16 * 25) return;               // wave-uniform guard
    int mt = tile / 25, nt = tile % 25;
    int m0 = mt * 16, n0 = nt * 16;

    v8f accj = {}; v8f acck = {};
    for (int k0 = 0; k0 < Kpad; k0 += 32) {
        v16h a  = load_fragA(A, Kpad, m0, k0, lane);
        v16h bj = load_fragB(W, Kpad, n0,        k0, lane);
        v16h bk = load_fragB(W, Kpad, n0 + HDIM, k0, lane);
        accj = __builtin_amdgcn_wmma_f32_16x16x32_f16(false, a, false, bj, (short)0, accj, false, false);
        acck = __builtin_amdgcn_wmma_f32_16x16x32_f16(false, a, false, bk, (short)0, acck, false, false);
    }
    int col   = n0 + (lane & 15);
    int rbase = m0 + ((lane & 16) ? 8 : 0);
    float bj = bias[col], bk = bias[HDIM + col];
#pragma unroll
    for (int v = 0; v < 8; ++v) {
        int row = rbase + v;
        float j  = sigm(accj[v] + bj);
        float kk = sigm(acck[v] + bk);
        float ho = h[(size_t)row * HDIM + col];
        h[(size_t)row * HDIM + col] = j * (1.0f - ho) + (1.0f - kk) * ho;
    }
}

// window pre-activation: kg(256x32) = h1pad @ winWt + winb
// grid: 8 blocks x 128 threads = 32 waves = 16(M) x 2(N) tiles
__global__ void win_gemm_kernel(const half_t* __restrict__ A, const half_t* __restrict__ W,
                                const float* __restrict__ wb, float* __restrict__ kg) {
    int lane = threadIdx.x & 31;
    int wave = threadIdx.x >> 5;
    int tile = blockIdx.x * 4 + wave;          // 0..31
    if (tile >= 32) return;                    // wave-uniform guard
    int mt = tile >> 1, nt = tile & 1;
    int m0 = mt * 16, n0 = nt * 16;
    v8f acc = {};
    for (int k0 = 0; k0 < K_W; k0 += 32) {
        v16h a  = load_fragA(A, K_W, m0, k0, lane);
        v16h bw = load_fragB(W, K_W, n0, k0, lane);
        acc = __builtin_amdgcn_wmma_f32_16x16x32_f16(false, a, false, bw, (short)0, acc, false, false);
    }
    int col   = n0 + (lane & 15);
    int rbase = m0 + ((lane & 16) ? 8 : 0);
    float bb = wb[col];
#pragma unroll
    for (int v = 0; v < 8; ++v)
        kg[(size_t)(rbase + v) * 32 + col] = acc[v] + bb;
}

// ---------------------------------------------------------------------------
// Attention window: alpha/beta/kappa from kg, phi over U, w = phi . c_vec
// one block (128 threads) per batch row
// ---------------------------------------------------------------------------
__global__ void attention_kernel(const float* __restrict__ kg, float* __restrict__ kappa,
                                 const float* __restrict__ c_vec, float* __restrict__ w) {
    int b   = blockIdx.x;
    int tid = threadIdx.x;   // 0..127
    __shared__ float alpha[KATT], beta[KATT], kap[KATT], phi[UDIM];
    if (tid < KATT) {
        float a  = expf(kg[(size_t)b * 32 + tid]);
        float be = expf(kg[(size_t)b * 32 + KATT + tid]);
        float kp = kappa[(size_t)b * KATT + tid] + expf(kg[(size_t)b * 32 + 2 * KATT + tid]);
        kappa[(size_t)b * KATT + tid] = kp;
        alpha[tid] = a; beta[tid] = be; kap[tid] = kp;
    }
    __syncthreads();
    float u = (float)tid;
    float p = 0.0f;
#pragma unroll
    for (int k = 0; k < KATT; ++k) {
        float d = kap[k] - u;
        p += alpha[k] * expf(-beta[k] * d * d);
    }
    phi[tid] = p;
    __syncthreads();
    const float* cv = c_vec + (size_t)b * UDIM * CDIM;
    float acc = 0.0f;
    for (int uu = 0; uu < UDIM; ++uu)
        acc += phi[uu] * cv[(size_t)uu * CDIM + tid];
    w[(size_t)b * CDIM + tid] = acc;
}

// ---------------------------------------------------------------------------
// Oscillator: 5 taps of r*cos(phi) with r,phi recurrence -> z (1280 x 416) f16
// ---------------------------------------------------------------------------
__global__ void osc_kernel(const float* __restrict__ sphi, const float* __restrict__ som,
                           const float* __restrict__ sr,  const float* __restrict__ smu,
                           half_t* __restrict__ zbuf) {
    int i = blockIdx.x * blockDim.x + threadIdx.x;  // 256*400
    if (i >= BATCH * HDIM) return;
    int b = i / HDIM, hc = i % HDIM;
    float r = sr[i], ph = sphi[i], mu = smu[i], om = som[i];
#pragma unroll
    for (int o = 0; o < 5; ++o) {
        if (o) { r = r + (mu - r * r) * r; ph = ph + om; }
        zbuf[(size_t)(b * 5 + o) * K_W + hc] = (half_t)(r * cosf(ph));
    }
}

// ---------------------------------------------------------------------------
// Final linear per step: out_rows(b*600 + t*5 + osc) = z @ linW + linb
// M=1280 (80 tiles), N=128 (8 tiles) -> 640 waves -> grid 160 x 128
// ---------------------------------------------------------------------------
__global__ void lin_gemm_kernel(const half_t* __restrict__ zbuf, const half_t* __restrict__ Wl,
                                const float* __restrict__ lb, float* __restrict__ out, int t) {
    int lane = threadIdx.x & 31;
    int wave = threadIdx.x >> 5;
    int tile = blockIdx.x * 4 + wave;          // 0..639
    if (tile >= 640) return;                   // wave-uniform guard
    int mt = tile >> 3, nt = tile & 7;
    int m0 = mt * 16, n0 = nt * 16;
    v8f acc = {};
    for (int k0 = 0; k0 < K_W; k0 += 32) {
        v16h a  = load_fragA(zbuf, K_W, m0, k0, lane);
        v16h bw = load_fragB(Wl,   K_W, n0, k0, lane);
        acc = __builtin_amdgcn_wmma_f32_16x16x32_f16(false, a, false, bw, (short)0, acc, false, false);
    }
    int col = n0 + (lane & 15);
    if (col < NOUT) {
        int rbase = m0 + ((lane & 16) ? 8 : 0);
        float bb = lb[col];
#pragma unroll
        for (int v = 0; v < 8; ++v) {
            int rl = rbase + v;                 // b*5 + osc
            int b = rl / 5, o = rl % 5;
            size_t grow = (size_t)b * 600 + (size_t)t * 5 + o;
            out[grow * NOUT + col] = acc[v] + bb;
        }
    }
}

// ---------------------------------------------------------------------------
// Host launch
// ---------------------------------------------------------------------------
extern "C" void kernel_launch(void* const* d_in, const int* in_sizes, int n_in,
                              void* d_out, int out_size, void* d_ws, size_t ws_size,
                              hipStream_t stream) {
    (void)in_sizes; (void)n_in; (void)out_size; (void)ws_size;

    // Input order: x, c_vec, then params dict flattened in insertion order:
    // rnn1,rnn2,phi,omega,r,mu (each: jxW,jxb,kxW,kxb,jhW,jhb,khW,khb),
    // then winW,winb,linW,linb.
    const float* x     = (const float*)d_in[0];
    const float* c_vec = (const float*)d_in[1];
    int idx = 2;
    struct FF { const float *jxW, *jxb, *kxW, *kxb, *jhW, *jhb, *khW, *khb; };
    auto getff = [&](void) {
        FF f;
        f.jxW = (const float*)d_in[idx++]; f.jxb = (const float*)d_in[idx++];
        f.kxW = (const float*)d_in[idx++]; f.kxb = (const float*)d_in[idx++];
        f.jhW = (const float*)d_in[idx++]; f.jhb = (const float*)d_in[idx++];
        f.khW = (const float*)d_in[idx++]; f.khb = (const float*)d_in[idx++];
        return f;
    };
    FF rnn1 = getff(), rnn2 = getff(), fphi = getff(), fom = getff(), fr = getff(), fmu = getff();
    const float* winW = (const float*)d_in[idx++];
    const float* winb = (const float*)d_in[idx++];
    const float* linW = (const float*)d_in[idx++];
    const float* linb = (const float*)d_in[idx++];
    float* out = (float*)d_out;

    // ---- carve workspace ----
    char* base = (char*)d_ws;
    size_t off = 0;
    auto carve = [&](size_t bytes) -> void* {
        off = (off + 255) & ~(size_t)255;
        void* p = base + off;
        off += bytes;
        return p;
    };
    half_t* W1p  = (half_t*)carve((size_t)800 * K_R1 * 2);
    half_t* W2p  = (half_t*)carve((size_t)800 * K_R2 * 2);
    half_t* Wsp[4];
    for (int i = 0; i < 4; ++i) Wsp[i] = (half_t*)carve((size_t)800 * K_SM * 2);
    half_t* Wwin = (half_t*)carve((size_t)32  * K_W * 2);
    half_t* Wlin = (half_t*)carve((size_t)128 * K_W * 2);
    half_t* A1   = (half_t*)carve((size_t)BATCH * K_R1 * 2);
    half_t* A2   = (half_t*)carve((size_t)BATCH * K_R2 * 2);
    half_t* As   = (half_t*)carve((size_t)BATCH * K_SM * 2);
    half_t* Aw   = (half_t*)carve((size_t)BATCH * K_W * 2);
    half_t* zbuf = (half_t*)carve((size_t)BATCH * 5 * K_W * 2);
    float* b1   = (float*)carve(800 * 4);
    float* b2   = (float*)carve(800 * 4);
    float* bs[4];
    for (int i = 0; i < 4; ++i) bs[i] = (float*)carve(800 * 4);
    float* bwin = (float*)carve(32 * 4);
    float* blin = (float*)carve(128 * 4);
    float* h1    = (float*)carve((size_t)BATCH * HDIM * 4);
    float* h2    = (float*)carve((size_t)BATCH * HDIM * 4);
    float* sphi  = (float*)carve((size_t)BATCH * HDIM * 4);
    float* som   = (float*)carve((size_t)BATCH * HDIM * 4);
    float* sr    = (float*)carve((size_t)BATCH * HDIM * 4);
    float* smu   = (float*)carve((size_t)BATCH * HDIM * 4);
    float* wvec  = (float*)carve((size_t)BATCH * CDIM * 4);
    float* kappa = (float*)carve((size_t)BATCH * KATT * 4);
    float* kg    = (float*)carve((size_t)BATCH * 32 * 4);

    // ---- zero everything we use (deterministic state) ----
    size_t n16 = (off + 15) / 16;
    zero_ws_kernel<<<(unsigned)((n16 + 255) / 256), 256, 0, stream>>>((uint4*)d_ws, n16);

    // ---- pack weights/biases (cheap, every call; no caching allowed) ----
    auto g1d = [](size_t n) { return (unsigned)((n + 255) / 256); };
    pack_ff_W_kernel<<<g1d((size_t)800 * K_R1), 256, 0, stream>>>(rnn1.jxW, rnn1.jhW, rnn1.kxW, rnn1.khW, 131, W1p, K_R1);
    pack_ff_W_kernel<<<g1d((size_t)800 * K_R2), 256, 0, stream>>>(rnn2.jxW, rnn2.jhW, rnn2.kxW, rnn2.khW, 531, W2p, K_R2);
    const FF* smalls[4] = { &fphi, &fom, &fr, &fmu };
    for (int i = 0; i < 4; ++i)
        pack_ff_W_kernel<<<g1d((size_t)800 * K_SM), 256, 0, stream>>>(smalls[i]->jxW, smalls[i]->jhW,
                                                                      smalls[i]->kxW, smalls[i]->khW,
                                                                      400, Wsp[i], K_SM);
    pack_ff_bias_kernel<<<g1d(800), 256, 0, stream>>>(rnn1.jxb, rnn1.jhb, rnn1.kxb, rnn1.khb, b1);
    pack_ff_bias_kernel<<<g1d(800), 256, 0, stream>>>(rnn2.jxb, rnn2.jhb, rnn2.kxb, rnn2.khb, b2);
    for (int i = 0; i < 4; ++i)
        pack_ff_bias_kernel<<<g1d(800), 256, 0, stream>>>(smalls[i]->jxb, smalls[i]->jhb,
                                                          smalls[i]->kxb, smalls[i]->khb, bs[i]);
    pack_win_W_kernel<<<g1d((size_t)32 * K_W), 256, 0, stream>>>(winW, Wwin);
    pack_lin_W_kernel<<<g1d((size_t)128 * K_W), 256, 0, stream>>>(linW, Wlin);
    pad_vec_kernel<<<1, 32, 0, stream>>>(winb, 30, bwin, 32);
    pad_vec_kernel<<<1, 128, 0, stream>>>(linb, NOUT, blin, 128);

    // ---- sequential scan over 120 downsampled steps ----
    float* sB[4] = { sphi, som, sr, smu };
    for (int t = 0; t < TSTEPS; ++t) {
        // h1 = ff(rnn1, [x_t, w], h1)
        assemble_A1_kernel<<<g1d((size_t)BATCH * K_R1), 256, 0, stream>>>(x, wvec, h1, A1, t);
        ff_gemm_kernel<<<100, 128, 0, stream>>>(A1, K_R1, W1p, b1, h1);
        // kg = h1 @ winW + winb ; attention -> kappa, w
        assemble_Aw_kernel<<<g1d((size_t)BATCH * K_W), 256, 0, stream>>>(h1, Aw);
        win_gemm_kernel<<<8, 128, 0, stream>>>(Aw, Wwin, bwin, kg);
        attention_kernel<<<BATCH, 128, 0, stream>>>(kg, kappa, c_vec, wvec);
        // h2 = ff(rnn2, [x_t, h1, w], h2)
        assemble_A2_kernel<<<g1d((size_t)BATCH * K_R2), 256, 0, stream>>>(x, h1, wvec, h2, A2, t);
        ff_gemm_kernel<<<100, 128, 0, stream>>>(A2, K_R2, W2p, b2, h2);
        // s_phi/s_omega/s_r/s_mu = ff(*, h2, s)
        for (int i = 0; i < 4; ++i) {
            assemble_As_kernel<<<g1d((size_t)BATCH * K_SM), 256, 0, stream>>>(h2, sB[i], As);
            ff_gemm_kernel<<<100, 128, 0, stream>>>(As, K_SM, Wsp[i], bs[i], sB[i]);
        }
        // oscillator taps -> z (f16), then final linear straight into out
        osc_kernel<<<g1d((size_t)BATCH * HDIM), 256, 0, stream>>>(sphi, som, sr, smu, zbuf);
        lin_gemm_kernel<<<160, 128, 0, stream>>>(zbuf, Wlin, blin, out, t);
    }
}
